// M3GNet_81475529605494
// MI455X (gfx1250) — matrix operand order repaired and checked by reading, not compiled, not run on hardware
//
#include <hip/hip_runtime.h>

// ---------------------------------------------------------------------------
// M3GNet forward for MI455X (gfx1250, wave32, WMMA).
// Dominant cost: per-edge gated MLPs ([E,192]@[192,64] chains) -> fused WMMA
// f16 kernels with f32 accumulation. Scatters via global f32 atomics.
// ---------------------------------------------------------------------------

#define NN      20000
#define NE      400000
#define NTRIP   1600000
#define NG      128

typedef __attribute__((ext_vector_type(16))) _Float16 v16h;
typedef __attribute__((ext_vector_type(8)))  float    v8f;

__device__ __forceinline__ float sigmf_(float x) { return 1.f / (1.f + __expf(-x)); }
__device__ __forceinline__ float siluf_(float x) { return x * sigmf_(x); }

// spherical Bessel j0..j3
__device__ __forceinline__ float bj0(float x) { return __sinf(x) / x; }
__device__ __forceinline__ float bj1(float x) { return __sinf(x) / (x * x) - __cosf(x) / x; }
__device__ __forceinline__ float bj2(float x) {
  float xi = 1.f / x;
  return (3.f * xi * xi * xi - xi) * __sinf(x) - 3.f * __cosf(x) * xi * xi;
}
__device__ __forceinline__ float bj3(float x) {
  float xi = 1.f / x, xi2 = xi * xi;
  return (15.f * xi2 * xi2 - 6.f * xi2) * __sinf(x) - (15.f * xi2 * xi - xi) * __cosf(x);
}

// rbf basis, l-major layout [l*3+n], CUTOFF=5
__device__ __forceinline__ void sbf9(float r, float* out) {
  const float factor = 0.12649110640673518f;  // sqrt(2/5^3)
  const float invc = 0.2f;
  const float r0[3] = {3.141592653589793f, 6.283185307179586f, 9.42477796076938f};
  const float r1[3] = {4.493409457909064f, 7.725251836937707f, 10.904121659428899f};
  const float r2[3] = {5.763459196894550f, 9.095011330476355f, 12.322940970566582f};
#pragma unroll
  for (int n = 0; n < 3; ++n) out[n]     = bj0(r * r0[n] * invc) * factor / fabsf(bj1(r0[n]));
#pragma unroll
  for (int n = 0; n < 3; ++n) out[3 + n] = bj1(r * r1[n] * invc) * factor / fabsf(bj2(r1[n]));
#pragma unroll
  for (int n = 0; n < 3; ++n) out[6 + n] = bj2(r * r2[n] * invc) * factor / fabsf(bj3(r2[n]));
}

// ---------------------------------------------------------------------------
// utility kernels
// ---------------------------------------------------------------------------
__global__ void zerof_kernel(float* p, long n) {
  long i = (long)blockIdx.x * 256 + threadIdx.x;
  if (i < n) p[i] = 0.f;
}
__global__ void copyf_kernel(float* d, const float* s, long n) {
  long i = (long)blockIdx.x * 256 + threadIdx.x;
  if (i < n) d[i] = s[i];
}

// per-(edge,col): bond_dist, rbf[E,9], tb_cut[E], edge_feat = silu(rbf@W+b)
__global__ void prep_edges_kernel(const float* __restrict__ bv, const float* __restrict__ w,
                                  const float* __restrict__ b, float* __restrict__ rbf,
                                  float* __restrict__ tbc, float* __restrict__ ef) {
  long i = (long)blockIdx.x * 256 + threadIdx.x;
  int e = (int)(i >> 6), c = (int)(i & 63);
  if (e >= NE) return;
  float vx = bv[(size_t)e * 3], vy = bv[(size_t)e * 3 + 1], vz = bv[(size_t)e * 3 + 2];
  float r = sqrtf(vx * vx + vy * vy + vz * vz);
  float s[9];
  sbf9(r, s);
  if (c < 9) rbf[(size_t)e * 9 + c] = s[c];
  if (c == 9) {  // poly_cutoff at TB_CUTOFF=4
    float ratio = r * 0.25f;
    float r3 = ratio * ratio * ratio;
    float val = 1.f - 6.f * r3 * ratio * ratio + 15.f * r3 * ratio - 10.f * r3;
    tbc[e] = (r <= 4.f) ? val : 0.f;
  }
  float acc = b[c];
#pragma unroll
  for (int k = 0; k < 9; ++k) acc += s[k] * w[k * 64 + c];
  ef[(size_t)e * 64 + c] = siluf_(acc);
}

__global__ void embed_kernel(const int* __restrict__ types, const float* __restrict__ tab,
                             float* __restrict__ node) {
  int i = blockIdx.x * 256 + threadIdx.x;
  if (i >= NN * 64) return;
  int n = i >> 6, c = i & 63;
  node[i] = tab[(size_t)types[n] * 64 + c];
}

// pack fp32 weights (din x 64) for one gated-MLP branch into WMMA f16 B-fragment
// layout: [B1: 6 chunks][B2: 2][B3: 2], each fragment = ((chunk*4+nt)*32+lane)*16+h
__global__ void pack_gmlp_kernel(const float* __restrict__ w1, const float* __restrict__ w2,
                                 const float* __restrict__ w3, _Float16* __restrict__ out) {
  int i = blockIdx.x * 256 + threadIdx.x;
  if (i >= 20480) return;
  const float* w;
  int idx;
  if (i < 12288)      { w = w1; idx = i; }
  else if (i < 16384) { w = w2; idx = i - 12288; }
  else                { w = w3; idx = i - 16384; }
  int h = idx & 15;
  int lane = (idx >> 4) & 31;
  int nt = (idx >> 9) & 3;
  int chunk = idx >> 11;
  int g = lane >> 4, n = lane & 15;
  int k = chunk * 32 + (h < 8 ? g * 8 + h : 16 + g * 8 + (h - 8));  // B layout: table order (k=8g+h for h<8, 16+8g+h-8 for h>=8)
  int col = nt * 16 + n;
  out[i] = (_Float16)w[k * 64 + col];
}

// atoms[N,9] = sigmoid(node @ Wa + ba)
__global__ void atoms_kernel(const float* __restrict__ node, const float* __restrict__ w,
                             const float* __restrict__ b, float* __restrict__ atoms) {
  int i = blockIdx.x * 256 + threadIdx.x;
  if (i >= NN * 9) return;
  int n = i / 9, j = i - n * 9;
  float acc = b[j];
  for (int d = 0; d < 64; ++d) acc += node[(size_t)n * 64 + d] * w[d * 9 + j];
  atoms[i] = sigmf_(acc);
}

// per triplet: basis = sbf(d2)*Y_l0(theta)*tbcut[a]*tbcut[b]*atoms[dst[b]]; scatter to l_src
__global__ void __launch_bounds__(256)
threebody_kernel(const float* __restrict__ bv, const float* __restrict__ tbc,
                 const int* __restrict__ ls, const int* __restrict__ ld,
                 const int* __restrict__ dst, const float* __restrict__ atoms,
                 float* __restrict__ nb) {
  int t = blockIdx.x * 256 + threadIdx.x;
  if (t >= NTRIP) return;
  int a = ls[t], b = ld[t];
  float ax = bv[(size_t)a * 3], ay = bv[(size_t)a * 3 + 1], az = bv[(size_t)a * 3 + 2];
  float bx = bv[(size_t)b * 3], by = bv[(size_t)b * 3 + 1], bz = bv[(size_t)b * 3 + 2];
  float d1 = sqrtf(ax * ax + ay * ay + az * az);
  float d2 = sqrtf(bx * bx + by * by + bz * bz);
  float ct = (ax * bx + ay * by + az * bz) / (d1 * d2);
  ct = fminf(1.f, fmaxf(-1.f, ct));
  float s[9];
  sbf9(d2, s);
  float y[3];
  y[0] = 0.28209479177387814f;
  y[1] = 0.4886025119029199f * ct;
  y[2] = 0.31539156525252005f * (3.f * ct * ct - 1.f);
  float w = tbc[a] * tbc[b];
  int ea = dst[b];
#pragma unroll
  for (int j = 0; j < 9; ++j) {
    float v = s[j] * y[j / 3] * w * atoms[(size_t)ea * 9 + j];
    atomicAdd(&nb[(size_t)a * 9 + j], v);
  }
}

// edge_feat += silu(nb@bond_l) * sigmoid(nb@bond_g)
__global__ void bond_update_kernel(const float* __restrict__ nb, const float* __restrict__ bl,
                                   const float* __restrict__ bg, float* __restrict__ ef) {
  long i = (long)blockIdx.x * 256 + threadIdx.x;
  int e = (int)(i >> 6), c = (int)(i & 63);
  if (e >= NE) return;
  float v[9];
#pragma unroll
  for (int k = 0; k < 9; ++k) v[k] = nb[(size_t)e * 9 + k];
  float h = 0.f, g = 0.f;
#pragma unroll
  for (int k = 0; k < 9; ++k) { h += v[k] * bl[k * 64 + c]; g += v[k] * bg[k * 64 + c]; }
  ef[(size_t)e * 64 + c] += siluf_(h) * sigmf_(g);
}

// ---------------------------------------------------------------------------
// fused WMMA conv kernel: edge gated MLP + node gated MLP, 64 edges / block
// ---------------------------------------------------------------------------
#define CPAD 208   // 416B row stride (16B aligned) -> ds_load_b128 A-frag loads
#define HPAD 80

struct ConvArgs {
  const float* node_cur; float* node_next; float* edge_feat;
  const float* rbf; const int* src; const int* dst;
  const _Float16 *eL, *eG;
  const float *eB0, *eB1, *eB2, *eGB0, *eGB1, *eGB2, *eW;
  const _Float16 *nL, *nG;
  const float *nB0, *nB1, *nB2, *nGB0, *nGB1, *nGB2, *nW;
};

__device__ __forceinline__ v8f wmma16(v16h a, v16h b, v8f c) {
  return __builtin_amdgcn_wmma_f32_16x16x32_f16(false, a, false, b, (short)0, c, false, false);
}

// A-frag per ISA 16-bit A layout: lane%16=row; lane/16 selects K-half; pairs along VGPRs
__device__ __forceinline__ v16h load_afrag(const _Float16* lds, int stride, int row, int kb,
                                           int lh) {
  v16h a;
  const _Float16* p = lds + row * stride + kb + lh * 8;
#pragma unroll
  for (int h = 0; h < 8; ++h) a[h] = p[h];
#pragma unroll
  for (int h = 0; h < 8; ++h) a[8 + h] = p[16 + h];
  return a;
}

__device__ __forceinline__ v16h load_bfrag(const _Float16* bp, int chunk, int nt, int lane) {
  return *(const v16h*)(bp + ((((chunk << 2) + nt) << 5) + lane) * 16);
}

__global__ void __launch_bounds__(128) conv_kernel(ConvArgs Ar) {
  __shared__ _Float16 Alds[64 * CPAD];   // feat = [node[src] | node[dst] | edge]  (f16)
  __shared__ _Float16 Hlds[64 * HPAD];   // main-branch intermediates
  __shared__ _Float16 Glds[64 * HPAD];   // gate-branch intermediates
  __shared__ float rbfs[64 * 10];
  __shared__ int sIdx[64], dIdx[64];

  const int tid = threadIdx.x;
  const int e0 = blockIdx.x * 64;

  if (tid < 64) { sIdx[tid] = Ar.src[e0 + tid]; dIdx[tid] = Ar.dst[e0 + tid]; }
  __syncthreads();

  for (int i = tid; i < 64 * 64; i += 128) {
    int r = i >> 6, c = i & 63;
    Alds[r * CPAD + c]       = (_Float16)Ar.node_cur[(size_t)sIdx[r] * 64 + c];
    Alds[r * CPAD + 64 + c]  = (_Float16)Ar.node_cur[(size_t)dIdx[r] * 64 + c];
    Alds[r * CPAD + 128 + c] = (_Float16)Ar.edge_feat[(size_t)(e0 + r) * 64 + c];
  }
  for (int i = tid; i < 64 * 9; i += 128) {
    int r = i / 9, c = i - r * 9;
    rbfs[r * 10 + c] = Ar.rbf[(size_t)(e0 + r) * 9 + c];
  }
  __syncthreads();

  const int lane = tid & 31;
  const int m0 = (tid >> 5) * 16;   // wave-private 16-row tile
  const int lh = lane >> 4;
  const int l15 = lane & 15;
  const int arow = m0 + l15;

  for (int phase = 0; phase < 2; ++phase) {  // 0: edge MLP, 1: node MLP
    const _Float16* BH = phase ? Ar.nL : Ar.eL;
    const _Float16* BG = phase ? Ar.nG : Ar.eG;
    const float* bHp[3] = {phase ? Ar.nB0 : Ar.eB0, phase ? Ar.nB1 : Ar.eB1,
                           phase ? Ar.nB2 : Ar.eB2};
    const float* bGp[3] = {phase ? Ar.nGB0 : Ar.eGB0, phase ? Ar.nGB1 : Ar.eGB1,
                           phase ? Ar.nGB2 : Ar.eGB2};
    const float* Wr = phase ? Ar.nW : Ar.eW;

    v8f hAcc[4], gAcc[4];

    // ---- layer 1: K = 192, A from Alds (same A for both branches) ----
#pragma unroll
    for (int nt = 0; nt < 4; ++nt) {
      float bh = bHp[0][nt * 16 + l15], bg = bGp[0][nt * 16 + l15];
      v8f hc, gc;
#pragma unroll
      for (int r = 0; r < 8; ++r) { hc[r] = bh; gc[r] = bg; }
#pragma unroll
      for (int c = 0; c < 6; ++c) {
        v16h a = load_afrag(Alds, CPAD, arow, c * 32, lh);
        hc = wmma16(a, load_bfrag(BH, c, nt, lane), hc);
        gc = wmma16(a, load_bfrag(BG, c, nt, lane), gc);
      }
      hAcc[nt] = hc; gAcc[nt] = gc;
    }
#pragma unroll
    for (int nt = 0; nt < 4; ++nt)
#pragma unroll
      for (int r = 0; r < 8; ++r) {
        int row = m0 + lh * 8 + r, col = nt * 16 + l15;
        Hlds[row * HPAD + col] = (_Float16)siluf_(hAcc[nt][r]);
        Glds[row * HPAD + col] = (_Float16)siluf_(gAcc[nt][r]);
      }

    // ---- layer 2: K = 64, branch-specific A ----
#pragma unroll
    for (int nt = 0; nt < 4; ++nt) {
      float bh = bHp[1][nt * 16 + l15], bg = bGp[1][nt * 16 + l15];
      v8f hc, gc;
#pragma unroll
      for (int r = 0; r < 8; ++r) { hc[r] = bh; gc[r] = bg; }
#pragma unroll
      for (int c = 0; c < 2; ++c) {
        v16h ah = load_afrag(Hlds, HPAD, arow, c * 32, lh);
        v16h ag = load_afrag(Glds, HPAD, arow, c * 32, lh);
        hc = wmma16(ah, load_bfrag(BH + 12288, c, nt, lane), hc);
        gc = wmma16(ag, load_bfrag(BG + 12288, c, nt, lane), gc);
      }
      hAcc[nt] = hc; gAcc[nt] = gc;   // buffered: safe to overwrite LDS afterwards
    }
#pragma unroll
    for (int nt = 0; nt < 4; ++nt)
#pragma unroll
      for (int r = 0; r < 8; ++r) {
        int row = m0 + lh * 8 + r, col = nt * 16 + l15;
        Hlds[row * HPAD + col] = (_Float16)siluf_(hAcc[nt][r]);
        Glds[row * HPAD + col] = (_Float16)siluf_(gAcc[nt][r]);
      }

    // ---- layer 3: K = 64, combine silu(h)*sigmoid(g) in registers ----
#pragma unroll
    for (int nt = 0; nt < 4; ++nt) {
      float bh = bHp[2][nt * 16 + l15], bg = bGp[2][nt * 16 + l15];
      v8f hc, gc;
#pragma unroll
      for (int r = 0; r < 8; ++r) { hc[r] = bh; gc[r] = bg; }
#pragma unroll
      for (int c = 0; c < 2; ++c) {
        v16h ah = load_afrag(Hlds, HPAD, arow, c * 32, lh);
        v16h ag = load_afrag(Glds, HPAD, arow, c * 32, lh);
        hc = wmma16(ah, load_bfrag(BH + 16384, c, nt, lane), hc);
        gc = wmma16(ag, load_bfrag(BG + 16384, c, nt, lane), gc);
      }
      hAcc[nt] = hc; gAcc[nt] = gc;
    }

    // ---- finalize: * (rbf @ Wr), then edge update or node-message scatter ----
#pragma unroll
    for (int nt = 0; nt < 4; ++nt)
#pragma unroll
      for (int r = 0; r < 8; ++r) {
        int row = m0 + lh * 8 + r, col = nt * 16 + l15;
        float h = siluf_(hAcc[nt][r]);
        float g = sigmf_(gAcc[nt][r]);
        float wsum = 0.f;
#pragma unroll
        for (int k = 0; k < 9; ++k) wsum += rbfs[row * 10 + k] * Wr[k * 64 + col];
        float v = h * g * wsum;
        size_t e = (size_t)(e0 + row);
        if (phase == 0) {
          float ne = Ar.edge_feat[e * 64 + col] + v;
          Ar.edge_feat[e * 64 + col] = ne;
          Alds[row * CPAD + 128 + col] = (_Float16)ne;  // refresh feat for node MLP
        } else {
          atomicAdd(&Ar.node_next[(size_t)dIdx[row] * 64 + col], v);
        }
      }
  }
}

// ---------------------------------------------------------------------------
// readout
// ---------------------------------------------------------------------------
__global__ void __launch_bounds__(64)
readout_kernel(const float* __restrict__ node, const int* __restrict__ gids,
               const float* __restrict__ w1, const float* __restrict__ b1,
               const float* __restrict__ w2, const float* __restrict__ b2,
               const float* __restrict__ wg, const float* __restrict__ bg,
               float* __restrict__ gf) {
  __shared__ float x[64], a1[64];
  __shared__ float gate;
  int n = blockIdx.x, c = threadIdx.x;
  x[c] = node[(size_t)n * 64 + c];
  __syncthreads();
  float acc = b1[c];
  for (int k = 0; k < 64; ++k) acc += x[k] * w1[k * 64 + c];
  a1[c] = siluf_(acc);
  if (c == 0) {
    float ga = bg[0];
    for (int k = 0; k < 64; ++k) ga += x[k] * wg[k];
    gate = sigmf_(ga);
  }
  __syncthreads();
  acc = b2[c];
  for (int k = 0; k < 64; ++k) acc += a1[k] * w2[k * 64 + c];
  atomicAdd(&gf[(size_t)gids[n] * 64 + c], siluf_(acc) * gate);
}

__global__ void __launch_bounds__(64)
final_kernel(const float* __restrict__ gf, const float* __restrict__ w1,
             const float* __restrict__ b1, const float* __restrict__ w2,
             const float* __restrict__ b2, const float* __restrict__ w3,
             const float* __restrict__ b3, float* __restrict__ out) {
  __shared__ float x[64], a1[64];
  int g = blockIdx.x, c = threadIdx.x;
  x[c] = gf[(size_t)g * 64 + c];
  __syncthreads();
  float acc = b1[c];
  for (int k = 0; k < 64; ++k) acc += x[k] * w1[k * 64 + c];
  a1[c] = siluf_(acc);
  __syncthreads();
  acc = b2[c];
  for (int k = 0; k < 64; ++k) acc += a1[k] * w2[k * 64 + c];
  __syncthreads();
  x[c] = siluf_(acc);
  __syncthreads();
  if (c == 0) {
    float o = b3[0];
    for (int k = 0; k < 64; ++k) o += x[k] * w3[k];
    out[g] = o;  // activate_last = False
  }
}

// ---------------------------------------------------------------------------
// host orchestration
// ---------------------------------------------------------------------------
extern "C" void kernel_launch(void* const* d_in, const int* in_sizes, int n_in, void* d_out,
                              int out_size, void* d_ws, size_t ws_size, hipStream_t stream) {
  (void)in_sizes; (void)n_in; (void)out_size; (void)ws_size;

  const float* bond_vec = (const float*)d_in[0];
  const int* node_types = (const int*)d_in[1];
  const int* src        = (const int*)d_in[2];
  const int* dst        = (const int*)d_in[3];
  const int* l_src      = (const int*)d_in[4];
  const int* l_dst      = (const int*)d_in[5];
  const int* graph_ids  = (const int*)d_in[6];
  auto P = [&](int i) { return (const float*)d_in[i]; };
  // jax pytree flatten (dict keys sorted): params leaves start at 7.
  // per block (30 leaves): atom.b,atom.w,bond_g,bond_l, e_gates(b,w)x3, e_layers(b,w)x3,
  //                        e_weight, n_gates(b,w)x3, n_layers(b,w)x3, n_weight
  // then edge_embed(b,w)=97,98; final(b,w)x3=99..104; node_embed=105;
  // readout_gate(b,w)=106,107; readout_mlp(b,w)x2=108..111

  size_t off = 0;
  char* base_ws = (char*)d_ws;
  auto carve = [&](size_t bytes) -> char* {
    char* p = base_ws + off;
    off += (bytes + 255) & ~(size_t)255;
    return p;
  };
  float* rbf        = (float*)carve((size_t)NE * 9 * 4);
  float* tbcut      = (float*)carve((size_t)NE * 4);
  float* edge_feat  = (float*)carve((size_t)NE * 64 * 4);
  float* atoms      = (float*)carve((size_t)NN * 9 * 4);
  float* new_bonds  = (float*)carve((size_t)NE * 9 * 4);
  float* nodeA      = (float*)carve((size_t)NN * 64 * 4);
  float* nodeB      = (float*)carve((size_t)NN * 64 * 4);
  float* graph_feat = (float*)carve((size_t)NG * 64 * 4);
  _Float16* packs   = (_Float16*)carve((size_t)12 * 20480 * 2);

  // edge prep + node embedding
  prep_edges_kernel<<<(NE * 64 + 255) / 256, 256, 0, stream>>>(bond_vec, P(98), P(97), rbf,
                                                               tbcut, edge_feat);
  embed_kernel<<<(NN * 64 + 255) / 256, 256, 0, stream>>>(node_types, P(105), nodeA);

  // pack gated-MLP weights to f16 WMMA B-fragment layout (once)
  for (int blk = 0; blk < 3; ++blk) {
    int b = 7 + 30 * blk;
    pack_gmlp_kernel<<<80, 256, 0, stream>>>(P(b + 11), P(b + 13), P(b + 15),
                                             packs + (blk * 4 + 0) * 20480);  // e_layers
    pack_gmlp_kernel<<<80, 256, 0, stream>>>(P(b + 5), P(b + 7), P(b + 9),
                                             packs + (blk * 4 + 1) * 20480);  // e_gates
    pack_gmlp_kernel<<<80, 256, 0, stream>>>(P(b + 24), P(b + 26), P(b + 28),
                                             packs + (blk * 4 + 2) * 20480);  // n_layers
    pack_gmlp_kernel<<<80, 256, 0, stream>>>(P(b + 18), P(b + 20), P(b + 22),
                                             packs + (blk * 4 + 3) * 20480);  // n_gates
  }

  float* cur = nodeA;
  float* nxt = nodeB;
  for (int blk = 0; blk < 3; ++blk) {
    int b = 7 + 30 * blk;
    zerof_kernel<<<(NE * 9 + 255) / 256, 256, 0, stream>>>(new_bonds, (long)NE * 9);
    atoms_kernel<<<(NN * 9 + 255) / 256, 256, 0, stream>>>(cur, P(b + 1), P(b + 0), atoms);
    threebody_kernel<<<NTRIP / 256, 256, 0, stream>>>(bond_vec, tbcut, l_src, l_dst, dst,
                                                      atoms, new_bonds);
    bond_update_kernel<<<(NE * 64 + 255) / 256, 256, 0, stream>>>(new_bonds, P(b + 3),
                                                                  P(b + 2), edge_feat);
    copyf_kernel<<<(NN * 64 + 255) / 256, 256, 0, stream>>>(nxt, cur, (long)NN * 64);

    ConvArgs ca;
    ca.node_cur = cur; ca.node_next = nxt; ca.edge_feat = edge_feat;
    ca.rbf = rbf; ca.src = src; ca.dst = dst;
    ca.eL = packs + (blk * 4 + 0) * 20480;
    ca.eG = packs + (blk * 4 + 1) * 20480;
    ca.nL = packs + (blk * 4 + 2) * 20480;
    ca.nG = packs + (blk * 4 + 3) * 20480;
    ca.eB0 = P(b + 10); ca.eB1 = P(b + 12); ca.eB2 = P(b + 14);
    ca.eGB0 = P(b + 4); ca.eGB1 = P(b + 6); ca.eGB2 = P(b + 8);
    ca.eW = P(b + 16);
    ca.nB0 = P(b + 23); ca.nB1 = P(b + 25); ca.nB2 = P(b + 27);
    ca.nGB0 = P(b + 17); ca.nGB1 = P(b + 19); ca.nGB2 = P(b + 21);
    ca.nW = P(b + 29);
    conv_kernel<<<NE / 64, 128, 0, stream>>>(ca);

    float* t = cur; cur = nxt; nxt = t;
  }

  // readout
  zerof_kernel<<<(NG * 64 + 255) / 256, 256, 0, stream>>>(graph_feat, (long)NG * 64);
  readout_kernel<<<NN, 64, 0, stream>>>(cur, graph_ids, P(109), P(108), P(111), P(110),
                                        P(107), P(106), graph_feat);
  final_kernel<<<NG, 64, 0, stream>>>(graph_feat, P(100), P(99), P(102), P(101), P(104),
                                      P(103), (float*)d_out);
}
